// OriginalSTFT_75900662055106
// MI455X (gfx1250) — matrix operand order repaired and checked by reading, not compiled, not run on hardware
//
#include <hip/hip_runtime.h>
#include <math.h>

typedef __attribute__((ext_vector_type(16))) _Float16 v16h;
typedef __attribute__((ext_vector_type(8)))  _Float16 v8h;
typedef __attribute__((ext_vector_type(8)))  float    v8f;

// Problem constants (from the reference)
constexpr int N_FFT     = 512;
constexpr int HOP       = 128;
constexpr int WIN       = 512;
constexpr int BATCH     = 32;
constexpr int N_SAMPLES = 480000;
constexpr int PAD       = N_FFT / 2;          // 256
constexpr int N_FREQ    = N_FFT / 2 + 1;      // 257
constexpr int N_FRAMES  = N_SAMPLES / HOP + 1;// 3751

// Tiling
constexpr int WAVES_PER_BLOCK  = 4;
constexpr int THREADS          = WAVES_PER_BLOCK * 32;            // 128
constexpr int FRAMES_PER_BLOCK = WAVES_PER_BLOCK * 16;            // 64 output columns
constexpr int SEG              = (FRAMES_PER_BLOCK - 1) * HOP + WIN; // 8576 samples staged in LDS

__global__ __launch_bounds__(THREADS)
void stft_mag_wmma(const float* __restrict__ x,
                   const float* __restrict__ basis,   // (514, 512): rows 0..256 real, 257..513 imag
                   float* __restrict__ out)           // (32, 257, 3751)
{
    __shared__ __attribute__((aligned(32))) _Float16 sAr[16 * 512];  // real basis tile, row-major
    __shared__ __attribute__((aligned(32))) _Float16 sAi[16 * 512];  // imag basis tile
    __shared__ __attribute__((aligned(32))) _Float16 sX[SEG];        // f16 x segment (reflect-resolved)

    const int tid = threadIdx.x;
    const int t0  = blockIdx.x * FRAMES_PER_BLOCK;   // first frame (output column) of this block
    const int f0  = blockIdx.y * 16;                 // first frequency row of this block
    const int b   = blockIdx.z;                      // batch

    // ---- Stage A tiles: convert f32 basis rows -> f16, zero-pad rows f >= 257 ----
    for (int i = tid; i < 16 * 512; i += THREADS) {
        const int r = i >> 9;          // row within tile
        const int c = i & 511;         // K
        const int f = f0 + r;
        float vr = 0.f, vi = 0.f;
        if (f < N_FREQ) {
            vr = basis[f * 512 + c];
            vi = basis[(N_FREQ + f) * 512 + c];
        }
        sAr[i] = (_Float16)vr;
        sAi[i] = (_Float16)vi;
    }

    // ---- Stage x segment: reflect padding resolved here, f32 -> f16 ----
    const float* xb = x + (size_t)b * N_SAMPLES;
    const int seg_base = t0 * HOP - PAD;             // signed sample index of segment start
    for (int i = tid; i < SEG; i += THREADS) {
        int idx = seg_base + i;
        idx = (idx < 0) ? -idx : idx;                                  // left reflect
        idx = (idx >= N_SAMPLES) ? (2 * N_SAMPLES - 2 - idx) : idx;    // right reflect
        idx = (idx < 0) ? 0 : ((idx >= N_SAMPLES) ? (N_SAMPLES - 1) : idx); // safety clamp
        sX[i] = (_Float16)xb[idx];
    }
    __syncthreads();

    // ---- WMMA main loop: one 16x16 output tile per wave, K = 512 in 16 steps of 32 ----
    const int lane = tid & 31;
    const int wave = tid >> 5;
    const int nn   = lane & 15;   // output column within tile / B matrix N
    const int hh   = lane >> 4;   // lane-half: selects K sub-range per ISA VGPR layouts

    // A fragment addressing (16-bit A 16x32 layout): M = lane&15;
    // lanes 0-15 hold K {k0..k0+7, k0+16..k0+23}; lanes 16-31 hold K {k0+8.., k0+24..}
    const int arow  = nn * 512;
    // B fragment addressing (16-bit B 32x16 layout): N = lane&15;
    // same K order as A: lanes 0-15 hold K {k0..k0+7, k0+16..k0+23}; lanes 16-31 hold K {k0+8.., k0+24..}
    const int bbase = (wave * 16 + nn) * HOP;

    v8f accR = {};
    v8f accI = {};
    #pragma unroll
    for (int ks = 0; ks < 16; ++ks) {
        const int k0 = ks * 32;

        union { v16h v; v8h h[2]; } ar, ai;
        ar.h[0] = *(const v8h*)&sAr[arow + k0 + hh * 8];
        ar.h[1] = *(const v8h*)&sAr[arow + k0 + 16 + hh * 8];
        ai.h[0] = *(const v8h*)&sAi[arow + k0 + hh * 8];
        ai.h[1] = *(const v8h*)&sAi[arow + k0 + 16 + hh * 8];

        const v16h bv = __builtin_shufflevector(*(const v8h*)&sX[bbase + k0 + hh * 8], *(const v8h*)&sX[bbase + k0 + 16 + hh * 8], 0, 1, 2, 3, 4, 5, 6, 7, 8, 9, 10, 11, 12, 13, 14, 15);

        // D = A*B + C, f32 accumulate; args: (neg_a, A, neg_b, B, c_mod, C, reuse_a, reuse_b)
        accR = __builtin_amdgcn_wmma_f32_16x16x32_f16(false, ar.v, false, bv,
                                                      (short)0, accR, false, false);
        accI = __builtin_amdgcn_wmma_f32_16x16x32_f16(false, ai.v, false, bv,
                                                      (short)0, accI, false, false);
    }

    // ---- Epilogue: magnitude + guarded store.
    // f32 C/D layout: VGPR r -> M = r + 8*(lane>=16), N = lane&15.
    const int t = t0 + wave * 16 + nn;
    if (t < N_FRAMES) {
        #pragma unroll
        for (int r = 0; r < 8; ++r) {
            const int f = f0 + r + hh * 8;
            if (f < N_FREQ) {
                const float re = accR[r];
                const float im = accI[r];
                out[((size_t)b * N_FREQ + f) * N_FRAMES + t] = sqrtf(re * re + im * im);
            }
        }
    }
}

extern "C" void kernel_launch(void* const* d_in, const int* in_sizes, int n_in,
                              void* d_out, int out_size, void* d_ws, size_t ws_size,
                              hipStream_t stream) {
    (void)in_sizes; (void)n_in; (void)out_size; (void)d_ws; (void)ws_size;
    const float* x     = (const float*)d_in[0];   // (32, 480000) f32
    const float* basis = (const float*)d_in[1];   // (514, 1, 512) f32
    float* out         = (float*)d_out;           // (32, 257, 3751) f32

    dim3 grid((N_FRAMES + FRAMES_PER_BLOCK - 1) / FRAMES_PER_BLOCK,  // 59 frame-tiles
              (N_FREQ + 15) / 16,                                    // 17 freq-tiles
              BATCH);                                                // 32
    stft_mag_wmma<<<grid, THREADS, 0, stream>>>(x, basis, out);
}
